// MultiQueryAttention_9457517985811
// MI455X (gfx1250) — matrix operand order repaired and checked by reading, not compiled, not run on hardware
//
#include <hip/hip_runtime.h>

// Problem constants (reference: B=2, S=2048, D_MODEL=1024, H=16, D_K=64)
#define BB 2
#define SS 2048
#define DM 1024
#define NH 16
#define DK 64
#define MROWS (BB*SS)          // 4096 rows of x / Q / attn-out

typedef __attribute__((ext_vector_type(16))) __bf16 v16bf;
typedef __attribute__((ext_vector_type(8)))  __bf16 v8bf;
typedef __attribute__((ext_vector_type(8)))  float  v8f;
typedef __attribute__((ext_vector_type(4)))  int    v4i;

#if __has_builtin(__builtin_amdgcn_global_load_async_to_lds_b128) && \
    __has_builtin(__builtin_amdgcn_s_wait_asynccnt)
#define HAVE_ASYNC_LDS 1
#else
#define HAVE_ASYNC_LDS 0
#endif

// Generic -> address-space casts via integers (LDS offset = addr[31:0] per ISA).
// b128 async builtin takes (v4i addrspace(1)*, v4i addrspace(3)*, imm, imm).
#define GPTR(p) ((__attribute__((address_space(1))) v4i*)(unsigned long long)(p))
#define LPTR(p) ((__attribute__((address_space(3))) v4i*)(unsigned int)(unsigned long long)(p))

__device__ __forceinline__ unsigned short f2bf(float f) {
    unsigned int u = __float_as_uint(f);
    u += 0x7FFFu + ((u >> 16) & 1u);       // round-to-nearest-even
    return (unsigned short)(u >> 16);
}
__device__ __forceinline__ float bf2f(unsigned short h) {
    return __uint_as_float(((unsigned int)h) << 16);
}

// Assemble a 16x32 bf16 A-fragment: per lane 16 contiguous halfs at p.
__device__ __forceinline__ v16bf load_a_frag(const unsigned short* p) {
    v8bf lo = *(const v8bf*)p;
    v8bf hi = *(const v8bf*)(p + 8);
    v16bf a;
    #pragma unroll
    for (int i = 0; i < 8; ++i) { a[i] = lo[i]; a[8 + i] = hi[i]; }
    return a;
}

// ---------------------------------------------------------------- prep kernels

__global__ void k_cvt_f32_bf16(const float* __restrict__ src,
                               unsigned short* __restrict__ dst, int n) {
    int i = blockIdx.x * blockDim.x + threadIdx.x;
    if (i < n) dst[i] = f2bf(src[i]);
}

// W (K x N) f32 row-major  ->  Wt (N x K) bf16 row-major
__global__ void k_transpose_w(const float* __restrict__ src,
                              unsigned short* __restrict__ dst, int K, int N) {
    int i = blockIdx.x * blockDim.x + threadIdx.x;
    if (i < K * N) {
        int k = i / N, n = i % N;
        dst[n * K + k] = f2bf(src[i]);
    }
}

// V (B*S, 64) bf16 -> Vt (B, 64, S) bf16
__global__ void k_transpose_v(const unsigned short* __restrict__ src,
                              unsigned short* __restrict__ dst) {
    int i = blockIdx.x * blockDim.x + threadIdx.x;
    if (i < MROWS * DK) {
        int b = i / (SS * DK);
        int r = i % (SS * DK);
        int s = r / DK, d = r % DK;
        dst[(b * DK + d) * SS + s] = src[i];
    }
}

// RoPE in place on Q (B,S,H*64) and K (B,S,64), bf16
__global__ void k_rope(unsigned short* __restrict__ Q,
                       unsigned short* __restrict__ Kb) {
    const int total = BB * SS * (NH + 1) * 32;
    int tid = blockIdx.x * blockDim.x + threadIdx.x;
    if (tid >= total) return;
    int i   = tid & 31;          // pair index 0..31
    int row = tid >> 5;
    int h = row % (NH + 1);
    int s = (row / (NH + 1)) % SS;
    int b = row / ((NH + 1) * SS);
    unsigned short* p;
    if (h < NH) p = Q  + (size_t)(b * SS + s) * DM + h * DK + 2 * i;
    else        p = Kb + (size_t)(b * SS + s) * DK + 2 * i;
    float x1 = bf2f(p[0]);
    float x2 = bf2f(p[1]);
    float inv = __expf(-(float)i * (9.210340371976184f / 32.0f)); // 1e4^(-i/32)
    float ang = (float)s * inv;
    float sn, cs;
    __sincosf(ang, &sn, &cs);
    p[0] = f2bf(x1 * cs - x2 * sn);
    p[1] = f2bf(x2 * cs + x1 * sn);
}

// ---------------------------------------------------------------- WMMA GEMM
// C(M,N) = A(M,K) @ B(K,N); A bf16 row-major, Bt = B^T bf16 (N x K) row-major.
// Register-blocked: one 32(M) x 64(N) tile per wave = 8 accumulators,
// 8 v_wmma_f32_16x16x32_bf16 per 32-deep k-step (1.5 b128 loads per wmma).
template <bool OUT_BF16>
__global__ void __launch_bounds__(256)
k_gemm_bf16(const unsigned short* __restrict__ A,
            const unsigned short* __restrict__ Bt,
            void* __restrict__ Out, int M, int N, int K) {
    const int wave = threadIdx.x >> 5;
    const int lane = threadIdx.x & 31;
    const int half = lane >> 4;
    const int n    = lane & 15;
    const int w  = blockIdx.x * (blockDim.x >> 5) + wave;
    const int tN = N >> 6;                    // 64-col tiles
    const int mt = w / tN;                    // 32-row tile index
    const int nt = w % tN;
    if (mt >= (M >> 5)) return;

    const int base0 = half * 16;
    const unsigned short* pA0 = A + (size_t)(mt * 32 + n) * K;
    const unsigned short* pA1 = A + (size_t)(mt * 32 + 16 + n) * K;
    const unsigned short* pB  = Bt + (size_t)(nt * 64 + n) * K + half * 16;

    v8f acc[2][4] = {};
    for (int k = 0; k < K; k += 32) {
        // prefetch ahead (global_prefetch_b8)
        __builtin_prefetch(pA0 + k + 256, 0, 1);
        __builtin_prefetch(pB  + k + 256, 0, 1);
        v16bf a0 = load_a_frag(pA0 + k + base0);
        v16bf a1 = load_a_frag(pA1 + k + base0);
        v16bf b0 = *(const v16bf*)(pB + k);
        v16bf b1 = *(const v16bf*)(pB + (size_t)16 * K + k);
        v16bf b2 = *(const v16bf*)(pB + (size_t)32 * K + k);
        v16bf b3 = *(const v16bf*)(pB + (size_t)48 * K + k);
        acc[0][0] = __builtin_amdgcn_wmma_f32_16x16x32_bf16(false, a0, false, b0, (short)0, acc[0][0], false, false);
        acc[0][1] = __builtin_amdgcn_wmma_f32_16x16x32_bf16(false, a0, false, b1, (short)0, acc[0][1], false, false);
        acc[0][2] = __builtin_amdgcn_wmma_f32_16x16x32_bf16(false, a0, false, b2, (short)0, acc[0][2], false, false);
        acc[0][3] = __builtin_amdgcn_wmma_f32_16x16x32_bf16(false, a0, false, b3, (short)0, acc[0][3], false, false);
        acc[1][0] = __builtin_amdgcn_wmma_f32_16x16x32_bf16(false, a1, false, b0, (short)0, acc[1][0], false, false);
        acc[1][1] = __builtin_amdgcn_wmma_f32_16x16x32_bf16(false, a1, false, b1, (short)0, acc[1][1], false, false);
        acc[1][2] = __builtin_amdgcn_wmma_f32_16x16x32_bf16(false, a1, false, b2, (short)0, acc[1][2], false, false);
        acc[1][3] = __builtin_amdgcn_wmma_f32_16x16x32_bf16(false, a1, false, b3, (short)0, acc[1][3], false, false);
    }
    // C layout: VGPR r -> row (+ r + half*8), col = n within each 16x16 tile
    #pragma unroll
    for (int mi = 0; mi < 2; ++mi) {
        #pragma unroll
        for (int ni = 0; ni < 4; ++ni) {
            #pragma unroll
            for (int r = 0; r < 8; ++r) {
                int row = mt * 32 + mi * 16 + r + half * 8;
                int col = nt * 64 + ni * 16 + n;
                if (OUT_BF16)
                    ((unsigned short*)Out)[(size_t)row * N + col] = f2bf(acc[mi][ni][r]);
                else
                    ((float*)Out)[(size_t)row * N + col] = acc[mi][ni][r];
            }
        }
    }
}

// ---------------------------------------------------------------- attention
// Block = 256 threads = 8 waves = 8 consecutive 16-row q tiles of one (b,h).
// KV tiles of 32 staged once per block in LDS via async-to-LDS (ASYNCcnt),
// shared by all 8 waves. Online softmax; P re-layout via per-wave LDS bounce.
__global__ void __launch_bounds__(256)
k_attn(const unsigned short* __restrict__ Q,
       const unsigned short* __restrict__ Kb,
       const unsigned short* __restrict__ Vt,
       unsigned short* __restrict__ Oc) {
    __shared__ __align__(64) unsigned short ldsK[32 * DK];      // K rows jb..jb+31   (4 KB)
    __shared__ __align__(64) unsigned short ldsV[DK * 32];      // Vt rows d, 32 kv   (4 KB)
    __shared__ __align__(64) unsigned short ldsP[8][16 * 32];   // per-wave P bounce  (8 KB)

    const int tid  = threadIdx.x;
    const int wave = tid >> 5;
    const int lane = tid & 31;
    const int half = lane >> 4;
    const int n    = lane & 15;
    const int QB = SS / 128;                    // 16 q-blocks per (b,h)
    const int qb = blockIdx.x % QB;
    const int h  = (blockIdx.x / QB) % NH;
    const int b  = blockIdx.x / (QB * NH);
    const int qt = qb * 8 + wave;               // this wave's 16-row q tile
    const int base0 = half * 16;

    // Q A-fragments: a0 covers d 0..31, a1 covers d 32..63
    const unsigned short* qp = Q + (size_t)(b * SS + qt * 16 + n) * DM + h * DK;
    v16bf a0 = load_a_frag(qp + base0);
    v16bf a1 = load_a_frag(qp + 32 + base0);

    v8f o0 = {}, o1 = {}, o2 = {}, o3 = {};     // O accum, 16x64 f32
    float m[8], l[8];
    #pragma unroll
    for (int r = 0; r < 8; ++r) { m[r] = -1e30f; l[r] = 0.0f; }

    const float scale = 0.125f;                 // 1/sqrt(64)
    const int jmaxB = qb * 4 + 3;               // block-uniform causal KV bound

    for (int j = 0; j <= jmaxB; ++j) {
        const int jb = j * 32;

        // ---- stage K (32x64, contiguous 4KB) and Vt (64 rows x 32 kv) tiles
        {
            const unsigned short* gk = Kb + (size_t)(b * SS + jb) * DK + tid * 8;
            unsigned short*       lk = &ldsK[tid * 8];
            const unsigned short* gv = Vt + (size_t)(b * DK + (tid >> 2)) * SS + jb + (tid & 3) * 8;
            unsigned short*       lv = &ldsV[(tid >> 2) * 32 + (tid & 3) * 8];
#if HAVE_ASYNC_LDS
            __builtin_amdgcn_global_load_async_to_lds_b128(GPTR(gk), LPTR(lk), 0, 0);
            __builtin_amdgcn_global_load_async_to_lds_b128(GPTR(gv), LPTR(lv), 0, 0);
            __builtin_amdgcn_s_wait_asynccnt(0);
#else
            *(v8bf*)lk = *(const v8bf*)gk;
            *(v8bf*)lv = *(const v8bf*)gv;
#endif
            __syncthreads();
        }

        // ---- scores S = Q @ K^T for 32 kv cols (two 16x16 C tiles)
        const unsigned short* kp0 = &ldsK[(size_t)n * DK + half * 16];
        const unsigned short* kp1 = &ldsK[(size_t)(16 + n) * DK + half * 16];
        v16bf b00 = *(const v16bf*)(kp0);        // d 0..31
        v16bf b01 = *(const v16bf*)(kp0 + 32);   // d 32..63
        v16bf b10 = *(const v16bf*)(kp1);
        v16bf b11 = *(const v16bf*)(kp1 + 32);
        v8f s0 = {}, s1 = {};
        s0 = __builtin_amdgcn_wmma_f32_16x16x32_bf16(false, a0, false, b00, (short)0, s0, false, false);
        s0 = __builtin_amdgcn_wmma_f32_16x16x32_bf16(false, a1, false, b01, (short)0, s0, false, false);
        s1 = __builtin_amdgcn_wmma_f32_16x16x32_bf16(false, a0, false, b10, (short)0, s1, false, false);
        s1 = __builtin_amdgcn_wmma_f32_16x16x32_bf16(false, a1, false, b11, (short)0, s1, false, false);

        // ---- online softmax (row spans 16 lanes of one half; xor<=8 stays inside)
        #pragma unroll
        for (int r = 0; r < 8; ++r) {
            int qrow = qt * 16 + r + half * 8;
            float v0 = (jb + n      <= qrow) ? s0[r] * scale : -1e30f;
            float v1 = (jb + 16 + n <= qrow) ? s1[r] * scale : -1e30f;
            float mx = fmaxf(v0, v1);
            mx = fmaxf(mx, __shfl_xor(mx, 1, 32));
            mx = fmaxf(mx, __shfl_xor(mx, 2, 32));
            mx = fmaxf(mx, __shfl_xor(mx, 4, 32));
            mx = fmaxf(mx, __shfl_xor(mx, 8, 32));
            float mnew = fmaxf(m[r], mx);
            float al = __expf(m[r] - mnew);
            float p0 = __expf(v0 - mnew);
            float p1 = __expf(v1 - mnew);
            float rs = p0 + p1;
            rs += __shfl_xor(rs, 1, 32);
            rs += __shfl_xor(rs, 2, 32);
            rs += __shfl_xor(rs, 4, 32);
            rs += __shfl_xor(rs, 8, 32);
            l[r] = l[r] * al + rs;
            m[r] = mnew;
            o0[r] *= al; o1[r] *= al; o2[r] *= al; o3[r] *= al;
            // park P (C-layout) into this wave's LDS tile, bf16
            ldsP[wave][(r + half * 8) * 32 + n]      = f2bf(p0);
            ldsP[wave][(r + half * 8) * 32 + 16 + n] = f2bf(p1);
        }

        // ---- reload P in A-layout (LDS is in-order within a wave)
        v16bf pa = load_a_frag(&ldsP[wave][n * 32 + base0]);

        // ---- O += P @ V  (ldsV rows are contiguous in kv)
        v16bf vb0 = *(const v16bf*)(&ldsV[(0 * 16 + n) * 32 + half * 16]);
        v16bf vb1 = *(const v16bf*)(&ldsV[(1 * 16 + n) * 32 + half * 16]);
        v16bf vb2 = *(const v16bf*)(&ldsV[(2 * 16 + n) * 32 + half * 16]);
        v16bf vb3 = *(const v16bf*)(&ldsV[(3 * 16 + n) * 32 + half * 16]);
        o0 = __builtin_amdgcn_wmma_f32_16x16x32_bf16(false, pa, false, vb0, (short)0, o0, false, false);
        o1 = __builtin_amdgcn_wmma_f32_16x16x32_bf16(false, pa, false, vb1, (short)0, o1, false, false);
        o2 = __builtin_amdgcn_wmma_f32_16x16x32_bf16(false, pa, false, vb2, (short)0, o2, false, false);
        o3 = __builtin_amdgcn_wmma_f32_16x16x32_bf16(false, pa, false, vb3, (short)0, o3, false, false);

        __syncthreads();   // all waves done reading ldsK/ldsV before next stage
    }

    // ---- normalize and store attn output (bf16, head-interleaved (B,S,1024))
    #pragma unroll
    for (int r = 0; r < 8; ++r) {
        float inv = 1.0f / l[r];
        int row = qt * 16 + r + half * 8;
        unsigned short* op = Oc + (size_t)(b * SS + row) * DM + h * DK;
        op[0 * 16 + n] = f2bf(o0[r] * inv);
        op[1 * 16 + n] = f2bf(o1[r] * inv);
        op[2 * 16 + n] = f2bf(o2[r] * inv);
        op[3 * 16 + n] = f2bf(o3[r] * inv);
    }
}

// ---------------------------------------------------------------- launcher

extern "C" void kernel_launch(void* const* d_in, const int* in_sizes, int n_in,
                              void* d_out, int out_size, void* d_ws, size_t ws_size,
                              hipStream_t stream) {
    const float* x  = (const float*)d_in[0];
    const float* Wq = (const float*)d_in[1];
    const float* Wk = (const float*)d_in[2];
    const float* Wv = (const float*)d_in[3];
    const float* Wo = (const float*)d_in[4];
    float* out = (float*)d_out;

    // workspace carve-up (bf16 buffers, 256B aligned)
    char* ws = (char*)d_ws;
    size_t off = 0;
    auto alloc = [&](size_t elems) {
        unsigned short* p = (unsigned short*)(ws + off);
        off = (off + elems * sizeof(unsigned short) + 255) & ~(size_t)255;
        return p;
    };
    unsigned short* xbf  = alloc((size_t)MROWS * DM);   // x in bf16
    unsigned short* WqT  = alloc((size_t)DM * DM);      // Wq^T
    unsigned short* WkT  = alloc((size_t)DK * DM);      // Wk^T
    unsigned short* WvT  = alloc((size_t)DK * DM);      // Wv^T
    unsigned short* WoT  = alloc((size_t)DM * DM);      // Wo^T
    unsigned short* Qbf  = alloc((size_t)MROWS * DM);   // Q (post-RoPE)
    unsigned short* Kbf  = alloc((size_t)MROWS * DK);   // K (post-RoPE)
    unsigned short* Vbf  = alloc((size_t)MROWS * DK);   // V
    unsigned short* Vtb  = alloc((size_t)MROWS * DK);   // V^T (B,64,S)
    unsigned short* Att  = alloc((size_t)MROWS * DM);   // attention output
    (void)ws_size; (void)in_sizes; (void)n_in; (void)out_size;

    // 1) convert x to bf16
    {
        int nel = MROWS * DM;
        k_cvt_f32_bf16<<<(nel + 255) / 256, 256, 0, stream>>>(x, xbf, nel);
    }
    // 2) transpose + convert weights
    k_transpose_w<<<(DM * DM + 255) / 256, 256, 0, stream>>>(Wq, WqT, DM, DM);
    k_transpose_w<<<(DM * DK + 255) / 256, 256, 0, stream>>>(Wk, WkT, DM, DK);
    k_transpose_w<<<(DM * DK + 255) / 256, 256, 0, stream>>>(Wv, WvT, DM, DK);
    k_transpose_w<<<(DM * DM + 255) / 256, 256, 0, stream>>>(Wo, WoT, DM, DM);

    // 3) projection GEMMs (bf16 WMMA, f32 accum, bf16 out); 32x64 tile/wave
    {
        int waves = (MROWS / 32) * (DM / 64);   // 2048
        k_gemm_bf16<true><<<waves / 8, 256, 0, stream>>>(xbf, WqT, Qbf, MROWS, DM, DM);
    }
    {
        int waves = (MROWS / 32) * (DK / 64);   // 128
        k_gemm_bf16<true><<<waves / 8, 256, 0, stream>>>(xbf, WkT, Kbf, MROWS, DK, DM);
        k_gemm_bf16<true><<<waves / 8, 256, 0, stream>>>(xbf, WvT, Vbf, MROWS, DK, DM);
    }

    // 4) RoPE on Q and K
    {
        int nth = BB * SS * (NH + 1) * 32;
        k_rope<<<(nth + 255) / 256, 256, 0, stream>>>(Qbf, Kbf);
    }

    // 5) transpose V for the P@V B-fragments
    k_transpose_v<<<(MROWS * DK + 255) / 256, 256, 0, stream>>>(Vbf, Vtb);

    // 6) flash attention: 512 blocks, KV staged per-block via async-to-LDS
    {
        int blocks = BB * NH * (SS / 128);      // 512
        k_attn<<<blocks, 256, 0, stream>>>(Qbf, Kbf, Vtb, Att);
    }

    // 7) output projection -> f32
    {
        int waves = (MROWS / 32) * (DM / 64);
        k_gemm_bf16<false><<<waves / 8, 256, 0, stream>>>(Att, WoT, (void*)out, MROWS, DM, DM);
    }
}
